// GNNEncoder_14405320311450
// MI455X (gfx1250) — matrix operand order repaired and checked by reading, not compiled, not run on hardware
//
#include <hip/hip_runtime.h>
#include <hip/hip_bf16.h>
#include <math.h>

// Problem constants (fixed by the reference).
#define DD   128
#define NND  100000      // NC == NI
#define EE   300000
#define LL   3
#define INV_SQRT_D 0.08838834764831843f

typedef __bf16 bf16;
typedef __attribute__((ext_vector_type(16))) __bf16 v16bf;
typedef __attribute__((ext_vector_type(8)))  float  v8f;

union AFrag { v16bf v; bf16  e[16]; };
union CFrag { v8f   v; float e[8];  };

// ---------------------------------------------------------------------------
// GEMM: Y[N,128] = X[N,128] @ W[128,128] (+ bias), bf16 WMMA, f32 accumulate.
// 256 threads = 8 waves; wave w owns rows [blk*128 + 16w, +16), all 128 cols.
// K = 128 -> 4 WMMA 16x16x32 steps per 16x16 output tile; 8 col tiles.
// W is transposed into LDS as bf16 so B-fragment loads are contiguous.
// ---------------------------------------------------------------------------
__global__ __launch_bounds__(256) void k_gemm_bias(
    const float* __restrict__ X, const float* __restrict__ W,
    const float* __restrict__ bias, float* __restrict__ Y, int N)
{
  __shared__ bf16 Wt[DD * DD];                 // Wt[n*128+k] = W[k*128+n], 32 KB
  const int tid = threadIdx.x;
  for (int i = tid; i < DD * DD; i += 256) {
    int k = i >> 7, n = i & 127;
    Wt[n * DD + k] = (bf16)W[i];
  }
  __syncthreads();

  const int wave = tid >> 5;
  const int lane = tid & 31;
  const int row0 = blockIdx.x * 128 + wave * 16;
  if (row0 >= N) return;                       // N % 16 == 0, per-wave guard OK

  const int m  = lane & 15;                    // A row / C column within tile
  const int hi = lane >> 4;                    // lane group (K sub-block select)
  const float4* xr = (const float4*)(X + (size_t)(row0 + m) * DD);

  // A fragments for all 4 K-chunks. Per ISA layout: lanes 0-15 hold
  // K = kb..kb+7 and kb+16..kb+23 with kb = 32*kc; lanes 16-31 kb += 8.
  AFrag a[4];
#pragma unroll
  for (int kc = 0; kc < 4; ++kc) {
    const int kb = kc * 32 + hi * 8;
    float4 f0 = xr[(kb >> 2) + 0];
    float4 f1 = xr[(kb >> 2) + 1];
    float4 f2 = xr[((kb + 16) >> 2) + 0];
    float4 f3 = xr[((kb + 16) >> 2) + 1];
    a[kc].e[0]=(bf16)f0.x; a[kc].e[1]=(bf16)f0.y; a[kc].e[2]=(bf16)f0.z; a[kc].e[3]=(bf16)f0.w;
    a[kc].e[4]=(bf16)f1.x; a[kc].e[5]=(bf16)f1.y; a[kc].e[6]=(bf16)f1.z; a[kc].e[7]=(bf16)f1.w;
    a[kc].e[8]=(bf16)f2.x; a[kc].e[9]=(bf16)f2.y; a[kc].e[10]=(bf16)f2.z; a[kc].e[11]=(bf16)f2.w;
    a[kc].e[12]=(bf16)f3.x; a[kc].e[13]=(bf16)f3.y; a[kc].e[14]=(bf16)f3.z; a[kc].e[15]=(bf16)f3.w;
  }

#pragma unroll
  for (int nt = 0; nt < 8; ++nt) {
    const int n = nt * 16 + m;                 // output column (and B column)
    CFrag c;
#pragma unroll
    for (int r = 0; r < 8; ++r) c.e[r] = 0.0f;
#pragma unroll
    for (int kc = 0; kc < 4; ++kc) {
      // B 32x16: lanes 0-15 hold K = kc*32 + 0..7,16..23; lanes 16-31 K = +8..15,24..31.
      const bf16* wc = &Wt[n * DD + kc * 32 + hi * 8];
      AFrag b;
#pragma unroll
      for (int j = 0; j < 16; ++j) b.e[j] = wc[(j & 7) + (j >> 3) * 16];
      c.v = __builtin_amdgcn_wmma_f32_16x16x32_bf16(
          false, a[kc].v, false, b.v, (short)0, c.v, false, false);
    }
    const float bv = bias ? bias[n] : 0.0f;
    float* yb = Y + (size_t)row0 * DD + n;
#pragma unroll
    for (int r = 0; r < 8; ++r)
      yb[(size_t)((r + hi * 8) * DD)] = c.e[r] + bv;   // C: M=r+8*hi, N=n
  }
}

// ---------------------------------------------------------------------------
// Edge attention pipeline (one wave32 per edge; lane -> one float4 of a row).
// ---------------------------------------------------------------------------
__global__ __launch_bounds__(256) void k_edge_logits(
    const float* __restrict__ Q, const float* __restrict__ Kt,
    const int* __restrict__ src, const int* __restrict__ dst,
    const float* __restrict__ pr, float* __restrict__ logits, int n)
{
  int e = (blockIdx.x * 256 + threadIdx.x) >> 5;
  int lane = threadIdx.x & 31;
  if (e >= n) return;
  int s = src[e], d = dst[e];
  float4 q = ((const float4*)(Q  + (size_t)d * DD))[lane];
  float4 k = ((const float4*)(Kt + (size_t)s * DD))[lane];
  float p = q.x * k.x + q.y * k.y + q.z * k.z + q.w * k.w;
#pragma unroll
  for (int o = 16; o > 0; o >>= 1) p += __shfl_xor(p, o, 32);
  if (lane == 0) logits[e] = p * pr[0] * INV_SQRT_D;
}

__device__ __forceinline__ unsigned enc_f(float f) {
  unsigned u = __float_as_uint(f);
  return (u & 0x80000000u) ? ~u : (u | 0x80000000u);   // order-preserving key
}

__global__ void k_seg_max(const float* __restrict__ logits,
                          const int* __restrict__ dst,
                          unsigned* __restrict__ menc, int n)
{
  int i = blockIdx.x * 256 + threadIdx.x;
  if (i >= n) return;
  atomicMax(&menc[dst[i]], enc_f(logits[i]));
}

__global__ void k_final_m(unsigned* __restrict__ m, int n)
{
  int i = blockIdx.x * 256 + threadIdx.x;
  if (i >= n) return;
  unsigned k = m[i];
  float f = (k & 0x80000000u) ? __uint_as_float(k ^ 0x80000000u)
                              : __uint_as_float(~k);
  if (!isfinite(f)) f = 0.0f;                          // reference: non-finite -> 0
  ((float*)m)[i] = f;
}

__global__ void k_expsum(float* __restrict__ logits,
                         const int* __restrict__ dst,
                         const float* __restrict__ m,
                         float* __restrict__ s, int n)
{
  int i = blockIdx.x * 256 + threadIdx.x;
  if (i >= n) return;
  int d = dst[i];
  float ex = expf(logits[i] - m[d]);
  logits[i] = ex;                                      // reuse as exp buffer
  atomicAdd(&s[d], ex);
}

__global__ __launch_bounds__(256) void k_scatter(
    const float* __restrict__ Vt,
    const int* __restrict__ src, const int* __restrict__ dst,
    const float* __restrict__ eattn, const float* __restrict__ ssum,
    float* __restrict__ agg, int n)
{
  int e = (blockIdx.x * 256 + threadIdx.x) >> 5;
  int lane = threadIdx.x & 31;
  if (e >= n) return;
  int s = src[e], d = dst[e];
  float a = eattn[e] / (ssum[d] + 1e-16f);
  float4 v = ((const float4*)(Vt + (size_t)s * DD))[lane];
  float* ap = agg + (size_t)d * DD + lane * 4;
  atomicAdd(ap + 0, v.x * a);
  atomicAdd(ap + 1, v.y * a);
  atomicAdd(ap + 2, v.z * a);
  atomicAdd(ap + 3, v.w * a);
}

// ---------------------------------------------------------------------------
// Elementwise helpers.
// ---------------------------------------------------------------------------
__global__ void k_fill_f32(float* p, float v, int n)
{ int i = blockIdx.x * 256 + threadIdx.x; if (i < n) p[i] = v; }

__global__ void k_fill_u32(unsigned* p, unsigned v, int n)
{ int i = blockIdx.x * 256 + threadIdx.x; if (i < n) p[i] = v; }

__global__ void k_gelu(const float* __restrict__ x, float* __restrict__ y, int n)
{
  int i = blockIdx.x * 256 + threadIdx.x;
  if (i >= n) return;
  float v = x[i];
  y[i] = 0.5f * v * (1.0f + erff(v * 0.70710678118654752f));   // exact GELU
}

__global__ void k_combine(const float* __restrict__ o, const float* __restrict__ x,
                          const float* __restrict__ skip, float* __restrict__ out,
                          int n, int do_relu)
{
  int i = blockIdx.x * 256 + threadIdx.x;
  if (i >= n) return;
  float sg = 1.0f / (1.0f + __expf(-skip[0]));
  float r = sg * o[i] + (1.0f - sg) * x[i];
  if (do_relu) r = fmaxf(r, 0.0f);
  out[i] = r;
}

// ---------------------------------------------------------------------------
// Orchestration.
// ---------------------------------------------------------------------------
extern "C" void kernel_launch(void* const* d_in, const int* in_sizes, int n_in,
                              void* d_out, int out_size, void* d_ws, size_t ws_size,
                              hipStream_t stream)
{
  (void)in_sizes; (void)n_in; (void)out_size; (void)ws_size;

  const float* x_code   = (const float*)d_in[0];
  const float* x_ident  = (const float*)d_in[1];
  const int*   ei_calls = (const int*)d_in[2];
  const int*   ei_has   = (const int*)d_in[3];
  const int*   ei_refs  = (const int*)d_in[4];
  const float* Wk_c = (const float*)d_in[5];
  const float* Wq_c = (const float*)d_in[6];
  const float* Wv_c = (const float*)d_in[7];
  const float* Wa_c = (const float*)d_in[8];
  const float* Wk_i = (const float*)d_in[9];
  const float* Wq_i = (const float*)d_in[10];
  const float* Wv_i = (const float*)d_in[11];
  const float* Wa_i = (const float*)d_in[12];
  const float* ar_calls = (const float*)d_in[13];
  const float* mr_calls = (const float*)d_in[14];
  const float* ar_has   = (const float*)d_in[15];
  const float* mr_has   = (const float*)d_in[16];
  const float* ar_refs  = (const float*)d_in[17];
  const float* mr_refs  = (const float*)d_in[18];
  const float* bk_c = (const float*)d_in[19];
  const float* bq_c = (const float*)d_in[20];
  const float* bv_c = (const float*)d_in[21];
  const float* ba_c = (const float*)d_in[22];
  const float* bk_i = (const float*)d_in[23];
  const float* bq_i = (const float*)d_in[24];
  const float* bv_i = (const float*)d_in[25];
  const float* ba_i = (const float*)d_in[26];
  const float* skip_c   = (const float*)d_in[27];
  const float* skip_i   = (const float*)d_in[28];
  const float* pr_calls = (const float*)d_in[29];
  const float* pr_has   = (const float*)d_in[30];
  const float* pr_refs  = (const float*)d_in[31];

  const size_t ND = (size_t)NND * DD;       // 12.8M floats per node buffer
  float* w = (float*)d_ws;
  float* Kc = w; w += ND;  float* Qc = w; w += ND;  float* Vc = w; w += ND;
  float* Ki = w; w += ND;  float* Qi = w; w += ND;  float* Vi = w; w += ND;
  float* kt = w; w += ND;  float* vt = w; w += ND;      // per-edge-type K'/V'
  float* aggc = w; w += ND;  float* aggi = w; w += ND;
  float* xcur_c = w; w += ND;  float* xcur_i = w; w += ND;
  float* logits = w; w += EE;
  float* mbuf = w; w += NND;                 // segment max (uint-encoded, then f32)
  float* sbuf = w; w += NND;                 // segment sum

  const dim3 B(256);
  const dim3 Ggemm((NND + 127) / 128);       // 782 blocks, 8 waves each
  const dim3 GedgeW((EE + 7) / 8);           // 1 wave per edge
  const dim3 GedgeT((EE + 255) / 256);
  const dim3 Gnode((NND + 255) / 256);
  const dim3 Gfeat(((int)ND + 255) / 256);

  const float* xc_p = x_code;
  const float* xi_p = x_ident;

  for (int l = 0; l < LL; ++l) {
    const size_t wo = (size_t)l * DD * DD;
    const size_t bo = (size_t)l * DD;

    // ---- node-level K/Q/V projections (6 WMMA GEMMs) ----
    k_gemm_bias<<<Ggemm, B, 0, stream>>>(xc_p, Wk_c + wo, bk_c + bo, Kc, NND);
    k_gemm_bias<<<Ggemm, B, 0, stream>>>(xc_p, Wq_c + wo, bq_c + bo, Qc, NND);
    k_gemm_bias<<<Ggemm, B, 0, stream>>>(xc_p, Wv_c + wo, bv_c + bo, Vc, NND);
    k_gemm_bias<<<Ggemm, B, 0, stream>>>(xi_p, Wk_i + wo, bk_i + bo, Ki, NND);
    k_gemm_bias<<<Ggemm, B, 0, stream>>>(xi_p, Wq_i + wo, bq_i + bo, Qi, NND);
    k_gemm_bias<<<Ggemm, B, 0, stream>>>(xi_p, Wv_i + wo, bv_i + bo, Vi, NND);

    k_fill_f32<<<Gfeat, B, 0, stream>>>(aggc, 0.0f, (int)ND);
    k_fill_f32<<<Gfeat, B, 0, stream>>>(aggi, 0.0f, (int)ND);

    // ---- three edge types: calls (c->c), refs (i->c), has (c->i) ----
    struct ET { const int* ei; const float* K; const float* V; const float* Q;
                const float* ar; const float* mr; const float* pr; float* agg; };
    const ET ets[3] = {
      { ei_calls, Kc, Vc, Qc, ar_calls + wo, mr_calls + wo, pr_calls + l, aggc },
      { ei_refs,  Ki, Vi, Qc, ar_refs  + wo, mr_refs  + wo, pr_refs  + l, aggc },
      { ei_has,   Kc, Vc, Qi, ar_has   + wo, mr_has   + wo, pr_has   + l, aggi },
    };
    for (int t = 0; t < 3; ++t) {
      const ET& e = ets[t];
      const int* src = e.ei;
      const int* dst = e.ei + EE;
      // relation transforms (2 WMMA GEMMs, no bias)
      k_gemm_bias<<<Ggemm, B, 0, stream>>>(e.K, e.ar, nullptr, kt, NND);
      k_gemm_bias<<<Ggemm, B, 0, stream>>>(e.V, e.mr, nullptr, vt, NND);
      // segment softmax over dst + weighted scatter
      k_fill_u32<<<Gnode, B, 0, stream>>>((unsigned*)mbuf, 0x007FFFFFu, NND); // enc(-inf)
      k_fill_f32<<<Gnode, B, 0, stream>>>(sbuf, 0.0f, NND);
      k_edge_logits<<<GedgeW, B, 0, stream>>>(e.Q, kt, src, dst, e.pr, logits, EE);
      k_seg_max<<<GedgeT, B, 0, stream>>>(logits, dst, (unsigned*)mbuf, EE);
      k_final_m<<<Gnode, B, 0, stream>>>((unsigned*)mbuf, NND);
      k_expsum<<<GedgeT, B, 0, stream>>>(logits, dst, mbuf, sbuf, EE);
      k_scatter<<<GedgeW, B, 0, stream>>>(vt, src, dst, logits, sbuf, e.agg, EE);
    }

    // ---- outputs: gelu -> Wa GEMM -> sigmoid-gated skip (+ relu on ident) ----
    float* out_c = (l == LL - 1) ? (float*)d_out : xcur_c;
    float* out_i = (l == LL - 1) ? ((float*)d_out + ND) : xcur_i;

    k_gelu<<<Gfeat, B, 0, stream>>>(aggc, kt, (int)ND);
    k_gemm_bias<<<Ggemm, B, 0, stream>>>(kt, Wa_c + wo, ba_c + bo, vt, NND);
    k_combine<<<Gfeat, B, 0, stream>>>(vt, xc_p, skip_c + l, out_c, (int)ND, 0);
    xc_p = out_c;

    k_gelu<<<Gfeat, B, 0, stream>>>(aggi, kt, (int)ND);
    k_gemm_bias<<<Ggemm, B, 0, stream>>>(kt, Wa_i + wo, ba_i + bo, vt, NND);
    k_combine<<<Gfeat, B, 0, stream>>>(vt, xi_p, skip_i + l, out_i, (int)ND, 1);
    xi_p = out_i;
  }
}
